// MultiHeadAttention_82248623719192
// MI455X (gfx1250) — matrix operand order repaired and checked by reading, not compiled, not run on hardware
//
#include <hip/hip_runtime.h>

// ---------------------------------------------------------------------------
// MI455X (gfx1250) multi-head attention forward, bf16 WMMA pipeline with
// async global->LDS staging (ASYNCcnt) in the flash-attention kernel.
// B=2, T=2048, D=1024, H=16, DH=64.
// ---------------------------------------------------------------------------

typedef __attribute__((ext_vector_type(16))) __bf16 v16bf;
typedef __attribute__((ext_vector_type(8)))  float  v8f;
typedef __attribute__((ext_vector_type(4)))  unsigned int v4u;

union Frag {            // one 16x32 A-fragment or 32x16 B-fragment (bf16)
    v16bf v;
    v4u   q[2];         // two 16-byte chunks
};

__device__ __forceinline__ v8f wmma_bf16(v16bf a, v16bf b, v8f c) {
    return __builtin_amdgcn_wmma_f32_16x16x32_bf16(
        false, a, false, b, (short)0, c, false, false);
}

// one per-lane 16-byte async copy: global -> LDS (tracked by ASYNCcnt)
__device__ __forceinline__ void async_copy_b128(const void* gptr, void* lptr) {
    const unsigned           lds = (unsigned)(unsigned long long)lptr;
    const unsigned long long ga  = (unsigned long long)gptr;
    asm volatile("global_load_async_to_lds_b128 %0, %1, off"
                 :: "v"(lds), "v"(ga) : "memory");
}
__device__ __forceinline__ void wait_asynccnt0() {
    asm volatile("s_wait_asynccnt 0x0" ::: "memory");
}

static constexpr int Bsz = 2;
static constexpr int T   = 2048;
static constexpr int D   = 1024;
static constexpr int H   = 16;
static constexpr int DH  = 64;
static constexpr int BH  = Bsz * H;       // 32
static constexpr int M   = Bsz * T;       // 4096 rows in GEMMs

// ---------------------------------------------------------------------------
// Elementwise converts
// ---------------------------------------------------------------------------
__global__ void cvt_f32_to_bf16(const float* __restrict__ in,
                                __bf16* __restrict__ out, int n) {
    int i = blockIdx.x * blockDim.x + threadIdx.x;
    if (i < n) out[i] = (__bf16)in[i];
}

__global__ void transpose_w_to_bf16(const float* __restrict__ W,
                                    __bf16* __restrict__ Wt) {
    int n = blockIdx.x * 16 + threadIdx.x;
    int k = blockIdx.y * 16 + threadIdx.y;
    Wt[(size_t)n * D + k] = (__bf16)W[(size_t)k * D + n];
}

// ---------------------------------------------------------------------------
// bf16 WMMA GEMM: C[M x D] = A[M x 1024] * B[1024 x D]
// Block 128 threads (4 waves); wave computes 32x64; double-buffered frags.
// MODE 0: bf16 -> [B,H,T,DH] ; MODE 1: bf16 -> [B,H,DH,T] ; MODE 2: f32 -> [M,D]
// ---------------------------------------------------------------------------
__device__ __forceinline__ void load_a2(Frag (&a)[2], const v4u* r0,
                                        const v4u* r1, int k0, int half) {
    a[0].q[0] = r0[(k0 + 8 * half) >> 3];
    a[0].q[1] = r0[(k0 + 16 + 8 * half) >> 3];
    a[1].q[0] = r1[(k0 + 8 * half) >> 3];
    a[1].q[1] = r1[(k0 + 16 + 8 * half) >> 3];
}

__device__ __forceinline__ void load_b4(Frag (&b)[4], const __bf16* Bt,
                                        int n0, int lm, int k0, int half) {
#pragma unroll
    for (int t = 0; t < 4; ++t) {
        const v4u* p = reinterpret_cast<const v4u*>(
            Bt + (size_t)(n0 + 16 * t + lm) * 1024 + k0);
        b[t].q[0] = p[half];
        b[t].q[1] = p[2 + half];
    }
}

__device__ __forceinline__ void mma_step(v8f (&acc)[2][4], const Frag (&a)[2],
                                         const Frag (&b)[4]) {
#pragma unroll
    for (int i = 0; i < 2; ++i)
#pragma unroll
        for (int t = 0; t < 4; ++t)
            acc[i][t] = wmma_bf16(a[i].v, b[t].v, acc[i][t]);
}

template <int MODE>
__global__ __launch_bounds__(128)
void gemm_bf16_wmma(const __bf16* __restrict__ A,
                    const __bf16* __restrict__ Bt,
                    void* __restrict__ dst) {
    const int lane = threadIdx.x & 31;
    const int wave = threadIdx.x >> 5;
    const int half = lane >> 4;
    const int lm   = lane & 15;

    const int m0 = blockIdx.y * 128 + wave * 32;
    const int n0 = blockIdx.x * 64;

    v8f acc[2][4];
    const v8f zero = {0.f, 0.f, 0.f, 0.f, 0.f, 0.f, 0.f, 0.f};
#pragma unroll
    for (int i = 0; i < 2; ++i)
#pragma unroll
        for (int t = 0; t < 4; ++t) acc[i][t] = zero;

    const v4u* ar0 = reinterpret_cast<const v4u*>(A + (size_t)(m0 + lm) * 1024);
    const v4u* ar1 = reinterpret_cast<const v4u*>(A + (size_t)(m0 + 16 + lm) * 1024);

    Frag a0[2], b0[4], a1[2], b1[4];
    load_a2(a0, ar0, ar1, 0, half);
    load_b4(b0, Bt, n0, lm, 0, half);

    for (int k0 = 0; k0 < 1024; k0 += 64) {
        load_a2(a1, ar0, ar1, k0 + 32, half);
        load_b4(b1, Bt, n0, lm, k0 + 32, half);
        mma_step(acc, a0, b0);
        if (k0 + 64 < 1024) {
            load_a2(a0, ar0, ar1, k0 + 64, half);
            load_b4(b0, Bt, n0, lm, k0 + 64, half);
        }
        mma_step(acc, a1, b1);
    }

#pragma unroll
    for (int i = 0; i < 2; ++i) {
#pragma unroll
        for (int t = 0; t < 4; ++t) {
#pragma unroll
            for (int r = 0; r < 8; ++r) {
                const int m = m0 + 16 * i + r + 8 * half;
                const int n = n0 + 16 * t + lm;
                const float val = acc[i][t][r];
                if (MODE == 2) {
                    ((float*)dst)[(size_t)m * D + n] = val;
                } else {
                    const int b  = m >> 11;
                    const int tt = m & 2047;
                    const int h  = n >> 6;
                    const int dh = n & 63;
                    size_t idx;
                    if (MODE == 0)
                        idx = (((size_t)(b * H + h) * T + tt) * DH + dh);
                    else
                        idx = (((size_t)(b * H + h) * DH + dh) * T + tt);
                    ((__bf16*)dst)[idx] = (__bf16)val;
                }
            }
        }
    }
}

// ---------------------------------------------------------------------------
// RoPE on Q and K, in place, layout [BH, T, DH], interleaved pairs.
// ---------------------------------------------------------------------------
__global__ void rope_kernel(__bf16* __restrict__ q, __bf16* __restrict__ k,
                            const float* __restrict__ rc,
                            const float* __restrict__ rs) {
    const int idx = blockIdx.x * blockDim.x + threadIdx.x;
    if (idx >= BH * T * (DH / 2)) return;
    const int j  = idx & 31;
    const int t  = (idx >> 5) & (T - 1);
    const int bh = idx >> 16;
    const float c = rc[t * 32 + j];
    const float s = rs[t * 32 + j];
    const size_t base = ((size_t)bh * T + t) * DH;

    {
        const float tr = (float)q[base + 2 * j];
        const float ti = (float)q[base + 2 * j + 1];
        q[base + 2 * j]     = (__bf16)(tr * c - ti * s);
        q[base + 2 * j + 1] = (__bf16)(tr * s + ti * c);
    }
    {
        const float tr = (float)k[base + 2 * j];
        const float ti = (float)k[base + 2 * j + 1];
        k[base + 2 * j]     = (__bf16)(tr * c - ti * s);
        k[base + 2 * j + 1] = (__bf16)(tr * s + ti * c);
    }
}

// ---------------------------------------------------------------------------
// Flash attention with async global->LDS K/V staging shared by 4 waves.
// Each wave owns one 16-query tile; the workgroup covers 64 adjacent queries
// and iterates a block-uniform causal bound so barriers stay uniform.
// q,k: [BH,T,DH] bf16 (post-RoPE).  vt: [BH,DH,T] bf16 (transposed).
// ---------------------------------------------------------------------------
static constexpr int SSTR = 36;   // f32  per S row   (144 B)
static constexpr int PSTR = 72;   // bf16 per P row   (144 B)
static constexpr int KSTR = 72;   // bf16 per K row   (144 B)  32 rows
static constexpr int VSTR = 40;   // bf16 per V row   ( 80 B)  64 rows

__global__ __launch_bounds__(128)
void flash_attn(const __bf16* __restrict__ q, const __bf16* __restrict__ k,
                const __bf16* __restrict__ vt, __bf16* __restrict__ attn) {
    __shared__ __align__(16) __bf16 sK[2][32 * KSTR];
    __shared__ __align__(16) __bf16 sV[2][64 * VSTR];
    __shared__ __align__(16) float  sS[4][16 * SSTR];
    __shared__ __align__(16) __bf16 sP[4][16 * PSTR];
    __shared__ __align__(16) float  sStat[4][16];

    const int tid  = threadIdx.x;
    const int lane = tid & 31;
    const int wave = tid >> 5;
    const int half = lane >> 4;
    const int lm   = lane & 15;

    const int bh    = blockIdx.y;
    const int qt    = blockIdx.x * 4 + wave;
    const int qbase = qt * 16;

    const __bf16* qh = q  + (size_t)bh * T * DH;
    const __bf16* kh = k  + (size_t)bh * T * DH;
    const __bf16* vh = vt + (size_t)bh * DH * T;

    float*  Srow = sS[wave];
    __bf16* Prow = sP[wave];
    float*  St   = sStat[wave];

    // stage one 32-key K/V block into LDS buffer `buf` (4 async b128 / thread)
    auto stage = [&](int jblk, int buf) {
        const int kb = jblk * 32;
#pragma unroll
        for (int s = 0; s < 2; ++s) {                 // K: 256 chunks of 16B
            const int c    = tid + 128 * s;
            const int key  = c >> 3;
            const int part = c & 7;
            async_copy_b128(kh + (size_t)(kb + key) * DH + part * 8,
                            sK[buf] + key * KSTR + part * 8);
        }
#pragma unroll
        for (int s = 0; s < 2; ++s) {                 // V: 256 chunks of 16B
            const int c    = tid + 128 * s;
            const int d    = c >> 2;
            const int part = c & 3;
            async_copy_b128(vh + (size_t)d * T + kb + part * 8,
                            sV[buf] + d * VSTR + part * 8);
        }
    };

    // Q A-fragments (registers, reused across all key blocks)
    Frag aQ[2];
    const v4u* qrow = reinterpret_cast<const v4u*>(qh + (size_t)(qbase + lm) * DH);
#pragma unroll
    for (int c = 0; c < 2; ++c) {
        aQ[c].q[0] = qrow[(32 * c + 8 * half) >> 3];
        aQ[c].q[1] = qrow[(32 * c + 16 + 8 * half) >> 3];
    }

    const v8f zero = {0.f, 0.f, 0.f, 0.f, 0.f, 0.f, 0.f, 0.f};
    v8f o[4];
#pragma unroll
    for (int t = 0; t < 4; ++t) o[t] = zero;

    const bool owner = (half == 0);       // lanes 0-15 own row lm
    float mrun = -__builtin_inff();
    float lrun = 0.f;

    // block-uniform causal bound (wave 3 has the longest range)
    const int   nblk  = (blockIdx.x * 4 + 3) / 2 + 1;
    const float scale = 0.125f;           // 1/sqrt(64)

    stage(0, 0);
    wait_asynccnt0();
    __syncthreads();

    for (int j = 0; j < nblk; ++j) {
        const int kb  = j * 32;
        const int buf = j & 1;

        if (j + 1 < nblk) stage(j + 1, buf ^ 1);   // overlap with compute

        // ---- S = Q * K^T from LDS ----
        Frag bK[2][2];
#pragma unroll
        for (int tile = 0; tile < 2; ++tile)
#pragma unroll
            for (int c = 0; c < 2; ++c) {
                const v4u* kr = reinterpret_cast<const v4u*>(
                    sK[buf] + (16 * tile + lm) * KSTR + 32 * c);
                bK[tile][c].q[0] = kr[half];
                bK[tile][c].q[1] = kr[2 + half];
            }
        v8f s0 = zero, s1 = zero;
        s0 = wmma_bf16(aQ[0].v, bK[0][0].v, s0);
        s0 = wmma_bf16(aQ[1].v, bK[0][1].v, s0);
        s1 = wmma_bf16(aQ[0].v, bK[1][0].v, s1);
        s1 = wmma_bf16(aQ[1].v, bK[1][1].v, s1);

        // ---- scale + causal mask, spill S tile to LDS (C layout) ----
#pragma unroll
        for (int r = 0; r < 8; ++r) {
            const int row = qbase + r + 8 * half;
            float v0 = s0[r] * scale;
            float v1 = s1[r] * scale;
            if (kb + lm > row)      v0 = -__builtin_inff();
            if (kb + 16 + lm > row) v1 = -__builtin_inff();
            Srow[(r + 8 * half) * SSTR + lm]      = v0;
            Srow[(r + 8 * half) * SSTR + 16 + lm] = v1;
        }

        // ---- per-row online softmax, lanes 0-15 only (row = lm) ----
        if (owner) {
            union { v4u u[8]; float f[32]; } sv;
            const v4u* sr = reinterpret_cast<const v4u*>(Srow + lm * SSTR);
#pragma unroll
            for (int i = 0; i < 8; ++i) sv.u[i] = sr[i];

            float mx = sv.f[0];
#pragma unroll
            for (int i = 1; i < 32; ++i) mx = fmaxf(mx, sv.f[i]);
            const float mnew  = fmaxf(mrun, mx);
            const float alpha = __expf(mrun - mnew);

            union { v4u u[4]; __bf16 h[32]; } pv;
            float sum = 0.f;
#pragma unroll
            for (int i = 0; i < 32; ++i) {
                const float p = __expf(sv.f[i] - mnew);
                sum += p;
                pv.h[i] = (__bf16)p;
            }
            lrun = lrun * alpha + sum;
            mrun = mnew;

            v4u* pw = reinterpret_cast<v4u*>(Prow + lm * PSTR);
#pragma unroll
            for (int i = 0; i < 4; ++i) pw[i] = pv.u[i];
            St[lm] = alpha;
        }

        // ---- broadcast alpha, rescale O ----
        union { v4u u[2]; float f[8]; } al;
        const v4u* ar = reinterpret_cast<const v4u*>(St + 8 * half);
        al.u[0] = ar[0];
        al.u[1] = ar[1];
#pragma unroll
        for (int t = 0; t < 4; ++t)
#pragma unroll
            for (int r = 0; r < 8; ++r) o[t][r] *= al.f[r];

        // ---- P A-fragment from LDS ----
        Frag aP;
        const v4u* pr = reinterpret_cast<const v4u*>(Prow + lm * PSTR);
        aP.q[0] = pr[half];
        aP.q[1] = pr[2 + half];

        // ---- O += P * V from LDS ----
        Frag bV[4];
#pragma unroll
        for (int t = 0; t < 4; ++t) {
            const v4u* vr = reinterpret_cast<const v4u*>(
                sV[buf] + (16 * t + lm) * VSTR);
            bV[t].q[0] = vr[half];
            bV[t].q[1] = vr[2 + half];
        }
#pragma unroll
        for (int t = 0; t < 4; ++t) o[t] = wmma_bf16(aP.v, bV[t].v, o[t]);

        wait_asynccnt0();     // next block staged (overlapped with the above)
        __syncthreads();
    }

    // ---- final 1/l normalization via LDS broadcast ----
    if (owner) St[lm] = 1.f / lrun;
    union { v4u u[2]; float f[8]; } li;
    {
        const v4u* lr = reinterpret_cast<const v4u*>(St + 8 * half);
        li.u[0] = lr[0];
        li.u[1] = lr[1];
    }

    const int b = bh >> 4, h = bh & 15;
#pragma unroll
    for (int t = 0; t < 4; ++t) {
#pragma unroll
        for (int r = 0; r < 8; ++r) {
            const int row = qbase + r + 8 * half;
            const int col = h * DH + 16 * t + lm;
            const float val = o[t][r] * li.f[r];
            attn[(size_t)(b * T + row) * D + col] = (__bf16)val;
        }
    }
}

// ---------------------------------------------------------------------------
// Launch
// ---------------------------------------------------------------------------
extern "C" void kernel_launch(void* const* d_in, const int* in_sizes, int n_in,
                              void* d_out, int out_size, void* d_ws, size_t ws_size,
                              hipStream_t stream) {
    const float* x   = (const float*)d_in[0];
    const float* Wq  = (const float*)d_in[1];
    const float* Wk  = (const float*)d_in[2];
    const float* Wv  = (const float*)d_in[3];
    const float* Wo  = (const float*)d_in[4];
    // d_in[5] = causal_mask (bool) — implemented analytically
    const float* rc  = (const float*)d_in[6];
    const float* rs  = (const float*)d_in[7];

    char* ws = (char*)d_ws;
    __bf16* xb    = (__bf16*)(ws);
    __bf16* Wqt   = (__bf16*)(ws + (8u << 20));
    __bf16* Wkt   = (__bf16*)(ws + (10u << 20));
    __bf16* Wvt   = (__bf16*)(ws + (12u << 20));
    __bf16* Wot   = (__bf16*)(ws + (14u << 20));
    __bf16* qb    = (__bf16*)(ws + (16u << 20));
    __bf16* kb    = (__bf16*)(ws + (24u << 20));
    __bf16* vtb   = (__bf16*)(ws + (32u << 20));
    __bf16* attnb = (__bf16*)(ws + (40u << 20));

    {
        const int n = M * D;
        cvt_f32_to_bf16<<<(n + 255) / 256, 256, 0, stream>>>(x, xb, n);
        dim3 g(D / 16, D / 16), blk(16, 16);
        transpose_w_to_bf16<<<g, blk, 0, stream>>>(Wq, Wqt);
        transpose_w_to_bf16<<<g, blk, 0, stream>>>(Wk, Wkt);
        transpose_w_to_bf16<<<g, blk, 0, stream>>>(Wv, Wvt);
        transpose_w_to_bf16<<<g, blk, 0, stream>>>(Wo, Wot);
    }
    {
        dim3 g(D / 64, M / 128), blk(128);
        gemm_bf16_wmma<0><<<g, blk, 0, stream>>>(xb, Wqt, (void*)qb);
        gemm_bf16_wmma<0><<<g, blk, 0, stream>>>(xb, Wkt, (void*)kb);
        gemm_bf16_wmma<1><<<g, blk, 0, stream>>>(xb, Wvt, (void*)vtb);
    }
    {
        const int n = BH * T * (DH / 2);
        rope_kernel<<<(n + 255) / 256, 256, 0, stream>>>(qb, kb, rc, rs);
    }
    {
        dim3 g((T / 16) / 4, BH), blk(128);
        flash_attn<<<g, blk, 0, stream>>>(qb, kb, vtb, attnb);
    }
    {
        dim3 g(D / 64, M / 128), blk(128);
        gemm_bf16_wmma<2><<<g, blk, 0, stream>>>(attnb, Wot, d_out);
    }
}
